// SelfLinearAttention_79559974191277
// MI455X (gfx1250) — compile-verified, matrix operand order checked by reading, not run on hardware
//
#include <hip/hip_runtime.h>

// ---------------- problem constants ----------------
#define B_   4
#define L_   16384
#define HS   192
#define E_   48
#define NH   4

#define TOK  32          // tokens per inner iteration
#define NIT1 4           // phase-1 iterations per block (128 tokens/block)
#define NIT2 4           // phase-2 iterations per block (128 tokens/block)
#define XSTR 264         // 4*64 + 8 halfs (528B stride, 16B-aligned)
#define KTSTR 40         // 32 + 8 halfs (80B stride)
#define ASTR 200         // 192 + 8 halfs (400B stride)

typedef __attribute__((ext_vector_type(16))) __bf16 v16bf;
typedef __attribute__((ext_vector_type(4)))  __bf16 v4bf;
typedef __attribute__((ext_vector_type(8)))  float  v8f;

union FragU { v16bf v; uint4 q[2]; };

// Load one 16x32 bf16 A/B fragment from a row-major [row][k] LDS tile.
// Layout (ISA 7.12.2, 16-bit A 16x32): lane L holds row (L&15);
// lanes 0-15: k in [0..7] then [16..23]; lanes 16-31: +8.
// => two 16-byte contiguous chunks per lane -> ds_load_b128 x2.
__device__ __forceinline__ v16bf ld_frag(const __bf16* tile, int rowstride,
                                         int row, int koff, int lane) {
  const uint4* p = (const uint4*)(tile + row * rowstride + koff + ((lane & 16) ? 8 : 0));
  FragU f;
  f.q[0] = p[0];   // k = kb + 0..7
  f.q[1] = p[2];   // k = kb + 16..23
  return f.v;
}

__device__ __forceinline__ v8f wmma_bf16(v16bf a, v16bf b, v8f c) {
  return __builtin_amdgcn_wmma_f32_16x16x32_bf16(false, a, false, b, (short)0, c,
                                                 false, false);
}

__device__ __forceinline__ v8f vzero() {
  v8f z = {0.f, 0.f, 0.f, 0.f, 0.f, 0.f, 0.f, 0.f};
  return z;
}

__device__ __forceinline__ v4bf cvt4(float4 f) {
  v4bf t;
  t[0] = (__bf16)f.x; t[1] = (__bf16)f.y; t[2] = (__bf16)f.z; t[3] = (__bf16)f.w;
  return t;
}

// ---------------- kv zero ----------------
__global__ void sla_zero(float* __restrict__ p, int n) {
  int i = blockIdx.x * blockDim.x + threadIdx.x;
  if (i < n) p[i] = 0.f;
}

// ---------------- phase 1: projections + norms + kv accumulation ----------------
__global__ __launch_bounds__(128)
void sla_phase1(const float* __restrict__ x,
                const float* __restrict__ Wq, const float* __restrict__ bq,
                const float* __restrict__ Wk, const float* __restrict__ bk,
                const float* __restrict__ Wv, const float* __restrict__ bv,
                __bf16* __restrict__ qws, float* __restrict__ kvg) {
  __shared__ __align__(16) __bf16 xs[TOK][XSTR];       // x tile, per-head K-padded
  __shared__ __align__(16) __bf16 wsh[3][E_][64];      // Wq/Wk/Wv, B-tile [n][k], zero-padded
  __shared__ __align__(16) __bf16 qout[TOK][ASTR];     // q staging for coalesced store
  __shared__ __align__(16) __bf16 kT[NH][E_][KTSTR];   // k transposed [e][t]
  __shared__ __align__(16) __bf16 vT[NH][E_][KTSTR];   // v transposed [e][t]

  const int tid  = threadIdx.x;
  const int lane = tid & 31;
  const int h    = tid >> 5;          // wave == head
  const int lm   = lane & 15;
  const int lh   = lane >> 4;
  const int bb   = blockIdx.y;

  // stage weights: valid region vectorized (float4 -> bf16x4)
  for (int idx = tid; idx < 3 * E_ * 12; idx += 128) {
    int w = idx / (E_ * 12);
    int r = (idx / 12) % E_;
    int c4 = (idx % 12) * 4;
    const float* W = (w == 0) ? Wq : ((w == 1) ? Wk : Wv);
    float4 f = *(const float4*)(W + r * E_ + c4);
    *(v4bf*)&wsh[w][r][c4] = cvt4(f);
  }
  // zero weight K-pad (c in 48..63)
  for (int idx = tid; idx < 3 * E_ * 16; idx += 128) {
    int w = idx / (E_ * 16);
    int r = (idx / 16) % E_;
    wsh[w][r][48 + (idx & 15)] = (__bf16)0.f;
  }
  // zero x padding (never overwritten by fills)
  for (int idx = tid; idx < TOK * XSTR; idx += 128) {
    int c = idx % XSTR;
    if (c >= 256 || (c & 63) >= E_) xs[idx / XSTR][c] = (__bf16)0.f;
  }
  // zero kT/vT padding columns (t >= 32)
  for (int idx = tid; idx < NH * E_ * KTSTR; idx += 128) {
    if ((idx % KTSTR) >= TOK) {
      ((__bf16*)kT)[idx] = (__bf16)0.f;
      ((__bf16*)vT)[idx] = (__bf16)0.f;
    }
  }

  float biasq[3], biask[3], biasv[3];
#pragma unroll
  for (int nt = 0; nt < 3; ++nt) {
    biasq[nt] = bq[nt * 16 + lm];
    biask[nt] = bk[nt * 16 + lm];
    biasv[nt] = bv[nt * 16 + lm];
  }

  v8f kvacc[3][3];
#pragma unroll
  for (int i = 0; i < 3; ++i)
#pragma unroll
    for (int j = 0; j < 3; ++j) kvacc[i][j] = vzero();

  __syncthreads();

  // hoist all weight B-fragments into registers (loop-invariant): 18 frags
  v16bf wf[3][3][2];   // [proj][nt][kstep]
#pragma unroll
  for (int w = 0; w < 3; ++w)
#pragma unroll
    for (int nt = 0; nt < 3; ++nt)
#pragma unroll
      for (int s = 0; s < 2; ++s)
        wf[w][nt][s] = ld_frag(&wsh[w][0][0], 64, nt * 16 + lm, 32 * s, lane);

  for (int it = 0; it < NIT1; ++it) {
    const int t0 = (blockIdx.x * NIT1 + it) * TOK;
    // stage x tile (float4 loads, bf16x4 LDS stores)
    for (int idx = tid; idx < TOK * (HS / 4); idx += 128) {
      int tt = idx / (HS / 4);
      int c4 = (idx % (HS / 4)) * 4;                 // multiple of 4, within a head
      float4 f = *(const float4*)(x + ((size_t)bb * L_ + t0 + tt) * HS + c4);
      *(v4bf*)&xs[tt][(c4 / E_) * 64 + (c4 % E_)] = cvt4(f);
    }
    // prefetch next tile (keep near) while computing this one
    if (it + 1 < NIT1) {
      const char* nx = (const char*)(x + ((size_t)bb * L_ + t0 + TOK) * HS);
      for (int idx = tid; idx < (TOK * HS * 4) / 256; idx += 128)
        __builtin_prefetch(nx + idx * 256, 0, 3);
    }
    __syncthreads();

#pragma unroll
    for (int mt = 0; mt < 2; ++mt) {
      v16bf a0 = ld_frag(&xs[0][0], XSTR, mt * 16 + lm, h * 64, lane);
      v16bf a1 = ld_frag(&xs[0][0], XSTR, mt * 16 + lm, h * 64 + 32, lane);

#pragma unroll
      for (int w = 0; w < 3; ++w) {   // 0:q 1:k 2:v
        v8f cq[3];
#pragma unroll
        for (int nt = 0; nt < 3; ++nt) cq[nt] = vzero();
        // K-step outer, N-tile inner: consecutive WMMAs share the A operand
#pragma unroll
        for (int nt = 0; nt < 3; ++nt) cq[nt] = wmma_bf16(a0, wf[w][nt][0], cq[nt]);
#pragma unroll
        for (int nt = 0; nt < 3; ++nt) cq[nt] = wmma_bf16(a1, wf[w][nt][1], cq[nt]);

        const float* bias = (w == 0) ? biasq : ((w == 1) ? biask : biasv);
#pragma unroll
        for (int nt = 0; nt < 3; ++nt)
#pragma unroll
          for (int r = 0; r < 8; ++r) cq[nt][r] += bias[nt];

#pragma unroll
        for (int r = 0; r < 8; ++r) {
          float s = cq[0][r] * cq[0][r] + cq[1][r] * cq[1][r] + cq[2][r] * cq[2][r];
          s += __shfl_xor(s, 1, 16); s += __shfl_xor(s, 2, 16);
          s += __shfl_xor(s, 4, 16); s += __shfl_xor(s, 8, 16);
          float sc = 1.f / fmaxf(sqrtf(s), 1e-12f);
          int tt = mt * 16 + r + 8 * lh;
#pragma unroll
          for (int nt = 0; nt < 3; ++nt) {
            float v = cq[nt][r] * sc;
            if (w == 0) {                      // q: fmap + staged global store
              v = (v > 0.f) ? (1.f + v) : __expf(v);
              qout[tt][h * E_ + nt * 16 + lm] = (__bf16)v;
            } else if (w == 1) {               // k: fmap, transposed [e][t]
              v = (v > 0.f) ? (1.f + v) : __expf(v);
              kT[h][nt * 16 + lm][tt] = (__bf16)v;
            } else {                            // v: no fmap, transposed [e][t]
              vT[h][nt * 16 + lm][tt] = (__bf16)v;
            }
          }
        }
      }
    }
    __syncthreads();

    // kv += k^T v  (contraction over 32 tokens, single K-step)
    {
      v16bf bf_[3];
#pragma unroll
      for (int ni = 0; ni < 3; ++ni)
        bf_[ni] = ld_frag(&vT[h][0][0], KTSTR, ni * 16 + lm, 0, lane);
#pragma unroll
      for (int mi = 0; mi < 3; ++mi) {
        v16bf a = ld_frag(&kT[h][0][0], KTSTR, mi * 16 + lm, 0, lane);
#pragma unroll
        for (int ni = 0; ni < 3; ++ni)
          kvacc[mi][ni] = wmma_bf16(a, bf_[ni], kvacc[mi][ni]);
      }
    }

    // coalesced q store: 128-bit copies (8 bf16 per op)
    for (int idx = tid; idx < TOK * (HS / 8); idx += 128) {
      int tt = idx / (HS / 8);
      int c8 = (idx % (HS / 8)) * 8;
      *(uint4*)(qws + ((size_t)bb * L_ + t0 + tt) * HS + c8) =
          *(const uint4*)&qout[tt][c8];
    }
    __syncthreads();
  }

  // flush kv accumulators
#pragma unroll
  for (int mi = 0; mi < 3; ++mi)
#pragma unroll
    for (int ni = 0; ni < 3; ++ni)
#pragma unroll
      for (int r = 0; r < 8; ++r) {
        int dq = mi * 16 + r + 8 * lh;
        int dv = ni * 16 + lm;
        atomicAdd(&kvg[(((size_t)bb * NH + h) * E_ + dq) * E_ + dv],
                  kvacc[mi][ni][r]);
      }
}

// ---------------- phase 2: attn = q@kv, out-proj, PReLU, LN, residual ----------------
__global__ __launch_bounds__(256)
void sla_phase2(const float* __restrict__ x, const __bf16* __restrict__ qws,
                const float* __restrict__ kvg, const float* __restrict__ Wp,
                const float* __restrict__ bpb, const float* __restrict__ prelu,
                const float* __restrict__ lng, const float* __restrict__ lnb,
                float* __restrict__ out) {
  __shared__ __align__(16) __bf16 wp[HS][ASTR];        // B-tile [e][k] == Wp row-major
  __shared__ __align__(16) __bf16 kvs[NH][E_][64];     // B-tile [dv][dq], K-padded
  __shared__ __align__(16) __bf16 qs[TOK][XSTR];       // q tile, per-head K-padded
  __shared__ __align__(16) __bf16 at[TOK][ASTR];       // attn tile [t][192]
  __shared__ __align__(16) float  ys[TOK][ASTR];       // pre-LN y tile

  const int tid  = threadIdx.x;
  const int lane = tid & 31;
  const int wave = tid >> 5;
  const int lm   = lane & 15;
  const int lh   = lane >> 4;
  const int h2   = wave & 3;   // head for attn step
  const int mh   = wave >> 2;  // token half for attn step
  const int mt   = wave & 1;        // out-proj M-tile for this wave
  const int ntb  = (wave >> 1) * 3; // out-proj N-tile base (3 consecutive)
  const int bb   = blockIdx.y;

  // stage Wp (float4 -> bf16x4)
  for (int idx = tid; idx < HS * (HS / 4); idx += 256) {
    int r = idx / (HS / 4);
    int c4 = (idx % (HS / 4)) * 4;
    float4 f = *(const float4*)(Wp + r * HS + c4);
    *(v4bf*)&wp[r][c4] = cvt4(f);
  }
  for (int idx = tid; idx < NH * E_ * 64; idx += 256) {
    int hh = idx / (E_ * 64);
    int dv = (idx >> 6) % E_;
    int dq = idx & 63;
    kvs[hh][dv][dq] = (dq < E_)
        ? (__bf16)kvg[(((size_t)bb * NH + hh) * E_ + dq) * E_ + dv]
        : (__bf16)0.f;
  }
  for (int idx = tid; idx < TOK * XSTR; idx += 256) {
    int c = idx % XSTR;
    if (c >= 256 || (c & 63) >= E_) qs[idx / XSTR][c] = (__bf16)0.f;
  }
  const float aprelu = prelu[0];
  float bpv[3];
#pragma unroll
  for (int j = 0; j < 3; ++j) bpv[j] = bpb[(ntb + j) * 16 + lm];
  __syncthreads();

  // hoist loop-invariant B fragments into registers
  v16bf kvf[3][2];     // attn B: kv for head h2
#pragma unroll
  for (int nt = 0; nt < 3; ++nt)
#pragma unroll
    for (int s = 0; s < 2; ++s)
      kvf[nt][s] = ld_frag(&kvs[h2][0][0], 64, nt * 16 + lm, 32 * s, lane);
  v16bf wpf[3][6];     // out-proj B: this wave's 3 N-tiles x 6 K-steps
#pragma unroll
  for (int j = 0; j < 3; ++j)
#pragma unroll
    for (int s = 0; s < 6; ++s)
      wpf[j][s] = ld_frag(&wp[0][0], ASTR, (ntb + j) * 16 + lm, 32 * s, lane);

  for (int it = 0; it < NIT2; ++it) {
    const int t0 = (blockIdx.x * NIT2 + it) * TOK;
    // stage q tile: 128-bit global loads -> 128-bit LDS stores (8 bf16 stay in-head)
    for (int idx = tid; idx < TOK * (HS / 8); idx += 256) {
      int tt = idx / (HS / 8);
      int c8 = (idx % (HS / 8)) * 8;
      *(uint4*)&qs[tt][(c8 / E_) * 64 + (c8 % E_)] =
          *(const uint4*)(qws + ((size_t)bb * L_ + t0 + tt) * HS + c8);
    }
    // prefetch next q tile and this tile's residual input (keep near)
    if (it + 1 < NIT2) {
      const char* nq = (const char*)(qws + ((size_t)bb * L_ + t0 + TOK) * HS);
      for (int idx = tid; idx < (TOK * HS * 2) / 256; idx += 256)
        __builtin_prefetch(nq + idx * 256, 0, 3);
    }
    {
      const char* nx = (const char*)(x + ((size_t)bb * L_ + t0) * HS);
      for (int idx = tid; idx < (TOK * HS * 4) / 256; idx += 256)
        __builtin_prefetch(nx + idx * 256, 0, 3);
    }
    __syncthreads();

    // attn = q @ kv : each wave -> (head h2, token half mh)
    {
      v16bf a0 = ld_frag(&qs[0][0], XSTR, mh * 16 + lm, h2 * 64, lane);
      v16bf a1 = ld_frag(&qs[0][0], XSTR, mh * 16 + lm, h2 * 64 + 32, lane);
      v8f ca[3];
#pragma unroll
      for (int nt = 0; nt < 3; ++nt) ca[nt] = vzero();
#pragma unroll
      for (int nt = 0; nt < 3; ++nt) ca[nt] = wmma_bf16(a0, kvf[nt][0], ca[nt]);
#pragma unroll
      for (int nt = 0; nt < 3; ++nt) ca[nt] = wmma_bf16(a1, kvf[nt][1], ca[nt]);
#pragma unroll
      for (int nt = 0; nt < 3; ++nt)
#pragma unroll
        for (int r = 0; r < 8; ++r)
          at[mh * 16 + r + 8 * lh][h2 * E_ + nt * 16 + lm] = (__bf16)ca[nt][r];
    }
    __syncthreads();

    // y = attn @ Wp^T + bp, PReLU : wave -> (mt, ntb..ntb+2); A shared per K-step
    {
      v8f cy[3];
#pragma unroll
      for (int j = 0; j < 3; ++j) cy[j] = vzero();
#pragma unroll
      for (int s = 0; s < 6; ++s) {
        v16bf a = ld_frag(&at[0][0], ASTR, mt * 16 + lm, 32 * s, lane);
#pragma unroll
        for (int j = 0; j < 3; ++j) cy[j] = wmma_bf16(a, wpf[j][s], cy[j]);
      }
#pragma unroll
      for (int j = 0; j < 3; ++j) {
        int n = (ntb + j) * 16 + lm;
#pragma unroll
        for (int r = 0; r < 8; ++r) {
          float v = cy[j][r] + bpv[j];
          v = (v >= 0.f) ? v : aprelu * v;
          ys[mt * 16 + r + 8 * lh][n] = v;
        }
      }
    }
    __syncthreads();

    // LayerNorm + residual: 8 lanes per token, all float4 traffic
    {
      int tt = tid >> 3;
      int g  = tid & 7;
      float s = 0.f, s2 = 0.f;
#pragma unroll
      for (int i = 0; i < 6; ++i) {
        float4 v4 = *(const float4*)&ys[tt][g * 24 + i * 4];
        s  += v4.x + v4.y + v4.z + v4.w;
        s2 += v4.x * v4.x + v4.y * v4.y + v4.z * v4.z + v4.w * v4.w;
      }
      s  += __shfl_xor(s, 1, 8);  s  += __shfl_xor(s, 2, 8);  s  += __shfl_xor(s, 4, 8);
      s2 += __shfl_xor(s2, 1, 8); s2 += __shfl_xor(s2, 2, 8); s2 += __shfl_xor(s2, 4, 8);
      float mu   = s * (1.f / 192.f);
      float var  = s2 * (1.f / 192.f) - mu * mu;
      float rstd = rsqrtf(var + 1e-6f);
      size_t base = ((size_t)bb * L_ + t0 + tt) * HS;
#pragma unroll
      for (int i = 0; i < 6; ++i) {
        int e = g * 24 + i * 4;
        float4 yv = *(const float4*)&ys[tt][e];
        float4 gv = *(const float4*)(lng + e);
        float4 bv = *(const float4*)(lnb + e);
        float4 xv = *(const float4*)(x + base + e);
        float4 o;
        o.x = xv.x + (yv.x - mu) * rstd * gv.x + bv.x;
        o.y = xv.y + (yv.y - mu) * rstd * gv.y + bv.y;
        o.z = xv.z + (yv.z - mu) * rstd * gv.z + bv.z;
        o.w = xv.w + (yv.w - mu) * rstd * gv.w + bv.w;
        *(float4*)(out + base + e) = o;
      }
    }
    __syncthreads();
  }
}

// ---------------- launcher ----------------
extern "C" void kernel_launch(void* const* d_in, const int* in_sizes, int n_in,
                              void* d_out, int out_size, void* d_ws, size_t ws_size,
                              hipStream_t stream) {
  const float* x  = (const float*)d_in[0];
  const float* Wq = (const float*)d_in[1];
  const float* bq = (const float*)d_in[2];
  const float* Wk = (const float*)d_in[3];
  const float* bk = (const float*)d_in[4];
  const float* Wv = (const float*)d_in[5];
  const float* bv = (const float*)d_in[6];
  const float* Wp = (const float*)d_in[7];
  const float* bp = (const float*)d_in[8];
  const float* pa = (const float*)d_in[9];
  const float* lg = (const float*)d_in[10];
  const float* lb = (const float*)d_in[11];
  float* out = (float*)d_out;

  __bf16* qws = (__bf16*)d_ws;                                    // 25,165,824 B
  float*  kvg = (float*)((char*)d_ws + (size_t)B_ * L_ * HS * 2); // 147,456 B

  const int nkv = B_ * NH * E_ * E_;
  sla_zero<<<(nkv + 255) / 256, 256, 0, stream>>>(kvg, nkv);

  dim3 g1(L_ / (TOK * NIT1), B_);
  sla_phase1<<<g1, 128, 0, stream>>>(x, Wq, bq, Wk, bk, Wv, bv, qws, kvg);

  dim3 g2(L_ / (TOK * NIT2), B_);
  sla_phase2<<<g2, 256, 0, stream>>>(x, qws, kvg, Wp, bp, pa, lg, lb, out);
}
